// VMLA_Block_2714419331913
// MI455X (gfx1250) — compile-verified, matrix operand order checked by reading, not run on hardware
//
#include <hip/hip_runtime.h>
#include <hip/hip_bf16.h>

// ---------------------------------------------------------------------------
// MI455X / gfx1250 implementation of the VMLA block.
// GEMMs: V_WMMA_F32_16X16X32_F16 (f16 in, fp32 accumulate), 32x64 per wave.
// NN GEMM stages its B tile in LDS via the Tensor Data Mover when available.
// Addressing: uniform SGPR base + 32-bit per-lane element offsets (GVS form)
// to minimize per-lane address VGPRs and avoid spills.
// ---------------------------------------------------------------------------

typedef __attribute__((ext_vector_type(16))) _Float16 v16h;
typedef __attribute__((ext_vector_type(8)))  float    v8f;
typedef __attribute__((ext_vector_type(4)))  unsigned u32x4;
typedef __attribute__((ext_vector_type(8)))  int      i32x8;
typedef __attribute__((ext_vector_type(4)))  int      i32x4;

#define WMMA_F16(a, b, c) \
  __builtin_amdgcn_wmma_f32_16x16x32_f16(false, (a), false, (b), (short)0, (c), false, false)

#if defined(__has_builtin)
#if __has_builtin(__builtin_amdgcn_tensor_load_to_lds) && \
    __has_builtin(__builtin_amdgcn_s_wait_tensorcnt)
#define HAVE_TDM 1
#else
#define HAVE_TDM 0
#endif
#else
#define HAVE_TDM 0
#endif

// Problem constants (from the reference).
constexpr int Bn   = 8;
constexpr int Ssz  = 1024;
constexpr int SRn  = 256;
constexpr int SNn  = 1024;
constexpr int D1n  = 768;
constexpr int D2n  = 512;
constexpr int D3n  = 768;
constexpr int MVHn = 256;
constexpr int Hn   = 12;
constexpr int HDn  = 64;   // per-head dim (32 content + 32 rotary)
constexpr int HHC  = 384;  // H * 32
constexpr int MLPn = 3072;

// LDS row stride for staged 32x64 tiles: 64 + 4 pad floats -> conflict-free
// column reads (k-halves land on disjoint bank groups).
constexpr int XPAD = 68;

// ---------------------------------------------------------------------------
// WMMA operand loaders (per CDNA5 ISA 16-bit A 16x32 layout):
// lane L holds row (L&15); half-elements i=0..15 map to
//   k = kb + (L<16 ? {0..7,16..23} : {8..15,24..31})
// B-operand (32x16) mirrors this with lane L holding column (L&15).
// base is wave-uniform; off is a 32-bit per-lane element offset.
// ---------------------------------------------------------------------------
__device__ __forceinline__ v16h load_row16u(const float* __restrict__ base,
                                            unsigned off, int kb) {
  int lane = threadIdx.x & 31;
  unsigned k0 = off + (unsigned)kb + ((lane & 16) ? 8u : 0u);
  unsigned k1 = off + (unsigned)kb + ((lane & 16) ? 24u : 16u);
  const float4* p0 = reinterpret_cast<const float4*>(base + k0);
  const float4* p1 = reinterpret_cast<const float4*>(base + k1);
  float4 x0 = p0[0], x1 = p0[1], y0 = p1[0], y1 = p1[1];
  v16h r;
  r[0]  = (_Float16)x0.x; r[1]  = (_Float16)x0.y; r[2]  = (_Float16)x0.z; r[3]  = (_Float16)x0.w;
  r[4]  = (_Float16)x1.x; r[5]  = (_Float16)x1.y; r[6]  = (_Float16)x1.z; r[7]  = (_Float16)x1.w;
  r[8]  = (_Float16)y0.x; r[9]  = (_Float16)y0.y; r[10] = (_Float16)y0.z; r[11] = (_Float16)y0.w;
  r[12] = (_Float16)y1.x; r[13] = (_Float16)y1.y; r[14] = (_Float16)y1.z; r[15] = (_Float16)y1.w;
  return r;
}

#if HAVE_TDM
// ---------------------------------------------------------------------------
// TDM descriptor for a 2D fp32 tile: 64 cols x 32 rows from a row-major
// tensor with row stride N (elements), DMA'd into LDS with 4-dword padding
// every 64 dwords (row stride 68 floats). CDNA5 ISA.
// ---------------------------------------------------------------------------
__device__ __forceinline__ void tdm_load_tile32x64(const float* gsrc,
                                                   unsigned lds_off, int N) {
  unsigned long long ga = (unsigned long long)(const void*)gsrc;
  u32x4 g0;
  g0[0] = 1u;                                  // count=1, user mode
  g0[1] = lds_off;                             // LDS byte address
  g0[2] = (unsigned)ga;                        // global_addr[31:0]
  g0[3] = (unsigned)((ga >> 32) & 0x01FFFFFFu) | (2u << 30);  // addr[56:32]|type=2
  i32x8 g1;
  g1[0] = (int)((2u << 16) |                   // data_size = 4B
                (1u << 20) |                   // pad_enable
                (5u << 22) |                   // pad_interval: 64 dwords
                (3u << 25));                   // pad_amount: 4 dwords
  g1[1] = (int)(64u << 16);                    // tensor_dim0[15:0] = 64
  g1[2] = (int)(32u << 16);                    // tensor_dim1[15:0] = 32
  g1[3] = (int)(64u << 16);                    // tile_dim0 = 64
  g1[4] = 32;                                  // tile_dim1 = 32, tile_dim2 = 0
  g1[5] = N;                                   // tensor_dim0_stride[31:0]
  g1[6] = 0;
  g1[7] = 0;
  i32x4 z4 = {0, 0, 0, 0};
#if __clang_major__ >= 23
  i32x8 z8 = {0, 0, 0, 0, 0, 0, 0, 0};
  __builtin_amdgcn_tensor_load_to_lds(g0, g1, z4, z4, z8, 0);
#else
  __builtin_amdgcn_tensor_load_to_lds(g0, g1, z4, z4, 0);
#endif
}
#endif

// ---------------------------------------------------------------------------
// GEMM NT:  C[z,M,N] = A[z,M,K] @ W[N,K]^T   (torch Linear, weight shared)
// block = 128 (4 waves); each wave computes a 32x64 strip (2 A tiles x 4 B
// tiles -> 8 WMMAs per k-step). All operand loads of a k-step are hoisted
// ahead of the WMMAs so the scheduler can clause them with staggered waits.
// grid(ceil(N/256), M/32, B)
// ---------------------------------------------------------------------------
__global__ void gemm_nt_wmma(const float* __restrict__ A, const float* __restrict__ W,
                             float* __restrict__ C, int M, int N, int K,
                             long sA, long sC) {
  int wave = threadIdx.x >> 5, lane = threadIdx.x & 31;
  int n0 = (blockIdx.x * 4 + wave) * 64;
  int m0 = blockIdx.y * 32;
  if (n0 >= N) return;
  const float* abase = A + (long)blockIdx.z * sA;   // uniform base
  unsigned aoff0 = (unsigned)(m0 + (lane & 15)) * (unsigned)K;
  unsigned aoff1 = aoff0 + 16u * (unsigned)K;
  unsigned woff  = (unsigned)(n0 + (lane & 15)) * (unsigned)K;
  unsigned wstep = 16u * (unsigned)K;
  v8f acc[2][4];
#pragma unroll
  for (int r = 0; r < 2; ++r)
#pragma unroll
    for (int j = 0; j < 4; ++j) acc[r][j] = v8f{};

  for (int kb = 0; kb < K; kb += 32) {
    v16h bv[4];
#pragma unroll
    for (int j = 0; j < 4; ++j) bv[j] = load_row16u(W, woff + (unsigned)j * wstep, kb);
    v16h a0 = load_row16u(abase, aoff0, kb);
    v16h a1 = load_row16u(abase, aoff1, kb);
#pragma unroll
    for (int j = 0; j < 4; ++j) acc[0][j] = WMMA_F16(a0, bv[j], acc[0][j]);
#pragma unroll
    for (int j = 0; j < 4; ++j) acc[1][j] = WMMA_F16(a1, bv[j], acc[1][j]);
  }
  float* cbase = C + (long)blockIdx.z * sC;         // uniform base
#pragma unroll
  for (int r = 0; r < 2; ++r) {
    int mo = m0 + r * 16 + ((lane & 16) ? 8 : 0);
#pragma unroll
    for (int j = 0; j < 4; ++j) {
      unsigned n = (unsigned)(n0 + j * 16 + (lane & 15));
#pragma unroll
      for (int v = 0; v < 8; ++v)
        cbase[(unsigned)(mo + v) * (unsigned)N + n] = acc[r][j][v];
    }
  }
}

// ---------------------------------------------------------------------------
// GEMM NN:  C[z,M,N] = Wt[M,K] @ X[z,K,N]    (seq-mix einsum, Wt shared)
// One wave per block, 32x64 strip. The 32x64 X tile is staged in LDS:
// TDM double-buffered (TENSORcnt) when available, coalesced copy otherwise.
// grid(N/64, M/32, B), block 32.
// ---------------------------------------------------------------------------
__global__ void gemm_nn_wmma(const float* __restrict__ Wt, const float* __restrict__ X,
                             float* __restrict__ C, int M, int N, int K,
                             long sX, long sC) {
  __shared__ float s_x[2][32 * XPAD];
  int lane = threadIdx.x & 31;
  int n0 = blockIdx.x * 64;
  int m0 = blockIdx.y * 32;
  const float* xb = X + (long)blockIdx.z * sX + n0; // uniform base
  unsigned aoff0 = (unsigned)(m0 + (lane & 15)) * (unsigned)K;
  unsigned aoff1 = aoff0 + 16u * (unsigned)K;
  int k0l = (lane & 16) ? 8 : 0;
  int k1l = (lane & 16) ? 24 : 16;
  v8f acc[2][4];
#pragma unroll
  for (int r = 0; r < 2; ++r)
#pragma unroll
    for (int j = 0; j < 4; ++j) acc[r][j] = v8f{};

#if HAVE_TDM
  tdm_load_tile32x64(xb, (unsigned)(unsigned long long)&s_x[0][0], N);
#endif
  for (int kb = 0; kb < K; kb += 32) {
#if HAVE_TDM
    int cur = (kb >> 5) & 1;
    if (kb + 32 < K) {
      tdm_load_tile32x64(xb + (long)(kb + 32) * N,
                         (unsigned)(unsigned long long)&s_x[cur ^ 1][0], N);
      __builtin_amdgcn_s_wait_tensorcnt(1);   // previous DMA complete
    } else {
      __builtin_amdgcn_s_wait_tensorcnt(0);
    }
    const float* sx = &s_x[cur][0];
#else
    const float* sx = &s_x[0][0];
#pragma unroll 4
    for (int i = lane; i < 32 * 64; i += 32) {
      int r = i >> 6, c = i & 63;
      s_x[0][r * XPAD + c] = xb[(unsigned)(kb + r) * (unsigned)N + (unsigned)c];
    }
#endif
    // B columns from LDS (read once, reused for both row tiles)
    v16h bv[4];
#pragma unroll
    for (int j = 0; j < 4; ++j) {
      int n = j * 16 + (lane & 15);
#pragma unroll
      for (int i = 0; i < 8; ++i) bv[j][i] = (_Float16)sx[(k0l + i) * XPAD + n];
#pragma unroll
      for (int i = 0; i < 8; ++i) bv[j][8 + i] = (_Float16)sx[(k1l + i) * XPAD + n];
    }
    v16h a0 = load_row16u(Wt, aoff0, kb);
    v16h a1 = load_row16u(Wt, aoff1, kb);
#pragma unroll
    for (int j = 0; j < 4; ++j) acc[0][j] = WMMA_F16(a0, bv[j], acc[0][j]);
#pragma unroll
    for (int j = 0; j < 4; ++j) acc[1][j] = WMMA_F16(a1, bv[j], acc[1][j]);
  }
  float* cbase = C + (long)blockIdx.z * sC;
#pragma unroll
  for (int r = 0; r < 2; ++r) {
    int mo = m0 + r * 16 + ((lane & 16) ? 8 : 0);
#pragma unroll
    for (int j = 0; j < 4; ++j) {
      unsigned n = (unsigned)(n0 + j * 16 + (lane & 15));
#pragma unroll
      for (int v = 0; v < 8; ++v)
        cbase[(unsigned)(mo + v) * (unsigned)N + n] = acc[r][j][v];
    }
  }
}

// ---------------------------------------------------------------------------
// LayerNorm (no bias): one 256-thread block per row of length D.
// ---------------------------------------------------------------------------
__global__ void ln_kernel(const float* __restrict__ x, const float* __restrict__ w,
                          float* __restrict__ y, int D) {
  long row = blockIdx.x;
  const float* xr = x + row * D;
  float* yr = y + row * D;
  float s = 0.f, ss = 0.f;
  for (int i = threadIdx.x; i < D; i += blockDim.x) {
    float v = xr[i];
    s += v; ss += v * v;
  }
#pragma unroll
  for (int off = 16; off; off >>= 1) {
    s  += __shfl_xor(s, off);
    ss += __shfl_xor(ss, off);
  }
  __shared__ float rs[8], rss[8];
  int wid = threadIdx.x >> 5;
  if ((threadIdx.x & 31) == 0) { rs[wid] = s; rss[wid] = ss; }
  __syncthreads();
  if (threadIdx.x == 0) {
    float a = 0.f, b = 0.f;
    for (int i = 0; i < (int)(blockDim.x >> 5); ++i) { a += rs[i]; b += rss[i]; }
    rs[0] = a; rss[0] = b;
  }
  __syncthreads();
  float mean = rs[0] / D;
  float var  = rss[0] / D - mean * mean;
  float inv  = rsqrtf(var + 1e-6f);
  for (int i = threadIdx.x; i < D; i += blockDim.x)
    yr[i] = (xr[i] - mean) * inv * w[i];
}

// ---------------------------------------------------------------------------
// z = mean + (e*eps + (1-e)*dirty) * exp(0.5*var);  dirty broadcast over batch
// ---------------------------------------------------------------------------
__global__ void reparam_kernel(const float* __restrict__ mean, const float* __restrict__ var,
                               const float* __restrict__ eps, const float* __restrict__ dirty,
                               const float* __restrict__ e2de, float* __restrict__ z,
                               long n, int pb) {
  long i = (long)blockIdx.x * blockDim.x + threadIdx.x;
  if (i >= n) return;
  float e = e2de[0];
  z[i] = mean[i] + (e * eps[i] + (1.f - e) * dirty[i % pb]) * expf(0.5f * var[i]);
}

// ---------------------------------------------------------------------------
// Build per-head Q/K [B,H,SN,64]: d<32 content part, d>=32 rotary part + RoPE
// ---------------------------------------------------------------------------
__global__ void rope_pack_kernel(const float* __restrict__ Xc, const float* __restrict__ Xr,
                                 const float* __restrict__ freqs, float* __restrict__ out) {
  long idx = (long)blockIdx.x * blockDim.x + threadIdx.x;
  long total = (long)Bn * Hn * SNn * HDn;
  if (idx >= total) return;
  int d  = (int)(idx & 63);
  int s  = (int)((idx >> 6) & (SNn - 1));
  int bh = (int)(idx >> 16);          // 6 (d) + 10 (s) bits
  int h  = bh % Hn;
  int b  = bh / Hn;
  unsigned rowc = (unsigned)(b * SNn + s) * (unsigned)HHC + (unsigned)(h * 32);
  float val;
  if (d < 32) {
    val = Xc[rowc + d];
  } else {
    int dr  = d - 32;
    float f = freqs[s * 16 + (dr & 15)];
    float x = Xr[rowc + dr];
    float rot = (dr < 16) ? -Xr[rowc + dr + 16] : Xr[rowc + dr - 16];
    val = x * cosf(f) + rot * sinf(f);
  }
  out[idx] = val;
}

// ---------------------------------------------------------------------------
// Fused attention: one wave per (b, h, 16-row query tile).
// Scores via WMMA (K=64 in two k=32 halves), online softmax through a
// wave-private LDS staging tile, P@V via WMMA over 32-KV tiles.
// V read from the [B,SN,768] projection; O written back in [B,SN,768].
// ---------------------------------------------------------------------------
__global__ void attn_kernel(const float* __restrict__ Q, const float* __restrict__ Kt,
                            const float* __restrict__ V, float* __restrict__ O) {
  __shared__ float    s_s[16][32];
  __shared__ _Float16 s_p[16][32];
  __shared__ float    s_m[16], s_l[16], s_alpha[16];

  int lane = threadIdx.x & 31;
  int b = blockIdx.z, h = blockIdx.y;
  int q0 = blockIdx.x * 16;
  unsigned bhBase = (unsigned)(b * Hn + h) * (unsigned)SNn;

  int nl = lane & 15;
  int mo = (lane & 16) ? 8 : 0;
  int k0 = (lane & 16) ? 8 : 0;
  int k1 = (lane & 16) ? 24 : 16;

  v16h aq0 = load_row16u(Q, (bhBase + (unsigned)(q0 + nl)) * (unsigned)HDn, 0);
  v16h aq1 = load_row16u(Q, (bhBase + (unsigned)(q0 + nl)) * (unsigned)HDn, 32);

  if (lane < 16) { s_m[lane] = -1e30f; s_l[lane] = 0.f; }

  v8f oacc[4];
#pragma unroll
  for (int t = 0; t < 4; ++t) oacc[t] = v8f{};

  const float scale = 0.125f;  // 1/sqrt(64)

  for (int kv0 = 0; kv0 < SNn; kv0 += 32) {
    // ---- scores: two 16x16 n-tiles over this 32-KV chunk
#pragma unroll
    for (int j = 0; j < 2; ++j) {
      unsigned koff = (bhBase + (unsigned)(kv0 + j * 16 + nl)) * (unsigned)HDn;
      v16h bk0 = load_row16u(Kt, koff, 0);
      v16h bk1 = load_row16u(Kt, koff, 32);
      v8f cs = {};
      cs = WMMA_F16(aq0, bk0, cs);
      cs = WMMA_F16(aq1, bk1, cs);
#pragma unroll
      for (int v = 0; v < 8; ++v) s_s[mo + v][j * 16 + nl] = cs[v] * scale;
    }
    // ---- online softmax, one row per lane 0..15 (single wave: lockstep)
    if (lane < 16) {
      int r = lane;
      float mold = s_m[r];
      float mt = mold;
#pragma unroll
      for (int c = 0; c < 32; ++c) mt = fmaxf(mt, s_s[r][c]);
      float alpha = expf(mold - mt);
      float lsum = 0.f;
#pragma unroll
      for (int c = 0; c < 32; ++c) {
        float p = expf(s_s[r][c] - mt);
        s_p[r][c] = (_Float16)p;
        lsum += p;
      }
      s_m[r] = mt;
      s_l[r] = s_l[r] * alpha + lsum;
      s_alpha[r] = alpha;
    }
    // ---- rescale accumulators by per-row alpha
#pragma unroll
    for (int t = 0; t < 4; ++t)
#pragma unroll
      for (int v = 0; v < 8; ++v) oacc[t][v] *= s_alpha[mo + v];

    // ---- P as A-operand (row = lane&15, k-mapped halves)
    v16h pa;
#pragma unroll
    for (int i = 0; i < 8; ++i) pa[i] = s_p[nl][k0 + i];
#pragma unroll
    for (int i = 0; i < 8; ++i) pa[8 + i] = s_p[nl][k1 + i];

    // ---- O += P @ V over four 16-wide d tiles
    unsigned vrow = (unsigned)(b * SNn + kv0);
#pragma unroll
    for (int t = 0; t < 4; ++t) {
      unsigned d = (unsigned)(h * HDn + t * 16 + nl);
      v16h bv;
#pragma unroll
      for (int i = 0; i < 8; ++i)
        bv[i] = (_Float16)V[(vrow + (unsigned)(k0 + i)) * (unsigned)D3n + d];
#pragma unroll
      for (int i = 0; i < 8; ++i)
        bv[8 + i] = (_Float16)V[(vrow + (unsigned)(k1 + i)) * (unsigned)D3n + d];
      oacc[t] = WMMA_F16(pa, bv, oacc[t]);
    }
  }

  // ---- normalize and store straight into [B,SN,768] layout
#pragma unroll
  for (int t = 0; t < 4; ++t) {
#pragma unroll
    for (int v = 0; v < 8; ++v) {
      float linv = 1.f / s_l[mo + v];
      O[(unsigned)(b * SNn + q0 + mo + v) * (unsigned)D3n +
        (unsigned)(h * HDn + t * 16 + nl)] = oacc[t][v] * linv;
    }
  }
}

// ---------------------------------------------------------------------------
// Elementwise helpers
// ---------------------------------------------------------------------------
__global__ void addto_kernel(float* __restrict__ y, const float* __restrict__ x, long n) {
  long i = (long)blockIdx.x * blockDim.x + threadIdx.x;
  if (i < n) y[i] += x[i];
}

__global__ void gelu_kernel(float* __restrict__ x, long n) {
  long i = (long)blockIdx.x * blockDim.x + threadIdx.x;
  if (i < n) {
    float v = x[i];
    x[i] = 0.5f * v * (1.f + erff(v * 0.70710678118654752f));
  }
}

// ---------------------------------------------------------------------------
// Host orchestration
// ---------------------------------------------------------------------------
extern "C" void kernel_launch(void* const* d_in, const int* in_sizes, int n_in,
                              void* d_out, int out_size, void* d_ws, size_t ws_size,
                              hipStream_t stream) {
  (void)in_sizes; (void)n_in; (void)out_size; (void)ws_size;
  const float* in_q    = (const float*)d_in[0];
  const float* ln_q_w  = (const float*)d_in[1];
  const float* ln_kv_w = (const float*)d_in[2];
  const float* Wt_mq   = (const float*)d_in[3];
  const float* Wt_vq   = (const float*)d_in[4];
  const float* Wt_mkv  = (const float*)d_in[5];
  const float* Wt_vkv  = (const float*)d_in[6];
  const float* mq_w1   = (const float*)d_in[7];
  const float* mq_w2   = (const float*)d_in[8];
  const float* vq_w1   = (const float*)d_in[9];
  const float* vq_w2   = (const float*)d_in[10];
  const float* mkv_w1  = (const float*)d_in[11];
  const float* mkv_w2  = (const float*)d_in[12];
  const float* vkv_w1  = (const float*)d_in[13];
  const float* vkv_w2  = (const float*)d_in[14];
  const float* dirty_zq  = (const float*)d_in[15];
  const float* dirty_zkv = (const float*)d_in[16];
  const float* e2de    = (const float*)d_in[17];
  const float* Wt_qzu  = (const float*)d_in[18];
  const float* Wt_kzu  = (const float*)d_in[19];
  const float* Wt_vzu  = (const float*)d_in[20];
  const float* Wt_qr   = (const float*)d_in[21];
  const float* Wt_kr   = (const float*)d_in[22];
  const float* qz_up   = (const float*)d_in[23];
  const float* kz_up   = (const float*)d_in[24];
  const float* vz_up   = (const float*)d_in[25];
  const float* q_w     = (const float*)d_in[26];
  const float* k_w     = (const float*)d_in[27];
  const float* v_w     = (const float*)d_in[28];
  const float* qr_w    = (const float*)d_in[29];
  const float* kr_w    = (const float*)d_in[30];
  const float* rope_qf = (const float*)d_in[31];
  const float* rope_kf = (const float*)d_in[32];
  const float* out_w   = (const float*)d_in[33];
  const float* ln2_w   = (const float*)d_in[34];
  const float* mlp_w1  = (const float*)d_in[35];
  const float* mlp_w2  = (const float*)d_in[36];
  const float* eps_zq  = (const float*)d_in[37];
  const float* eps_zkv = (const float*)d_in[38];
  float* out = (float*)d_out;
  float* ws  = (float*)d_ws;

  // ---- workspace layout (floats), with liveness-based reuse ----
  constexpr long SZ_SD1 = (long)Bn * Ssz * D1n;   // 6,291,456
  constexpr long SZ_T   = (long)Bn * SRn * D1n;   // 1,572,864
  constexpr long SZ_U   = (long)Bn * SRn * D2n;   // 1,048,576
  constexpr long SZ_Z   = (long)Bn * SRn * MVHn;  //   524,288
  constexpr long SZ_TU  = (long)Bn * SNn * MVHn;  // 2,097,152
  constexpr long SZ_QZ  = (long)Bn * SNn * D2n;   // 4,194,304
  constexpr long SZ_QC  = (long)Bn * SNn * HHC;   // 3,145,728
  constexpr long SZ_H   = (long)Bn * Hn * SNn * HDn; // 6,291,456

  long o = 0;
  float* XQ   = ws + o; o += SZ_SD1;
  float* XKV  = ws + o; o += SZ_SD1;
  float* T    = ws + o; o += SZ_T;
  float* U    = ws + o; o += SZ_U;
  float* MZQ  = ws + o; o += SZ_Z;
  float* VZQ  = ws + o; o += SZ_Z;
  float* MZKV = ws + o; o += SZ_Z;
  float* VZKV = ws + o; o += SZ_Z;
  float* ZQ   = ws + o; o += SZ_Z;
  float* ZKV  = ws + o; o += SZ_Z;
  float* TQZ  = ws + o; o += SZ_TU;
  float* TKZ  = ws + o; o += SZ_TU;
  float* TVZ  = ws + o; o += SZ_TU;
  float* TQR  = ws + o; o += SZ_TU;
  float* QZ   = ws + o; o += SZ_QZ;
  float* KZ   = ws + o; o += SZ_QZ;
  float* VZ   = ws + o; o += SZ_QZ;
  float* QC   = ws + o; o += SZ_QC;
  float* KC   = ws + o; o += SZ_QC;
  float* QR   = ws + o; o += SZ_QC;
  float* KR   = ws + o; o += SZ_QC;
  float* Qh   = ws + o; o += SZ_H;
  float* Kh   = ws + o; o += SZ_H;
  float* Obuf = ws + o; o += SZ_SD1;
  float* X    = ws + o; o += SZ_SD1;
  float* Y    = ws + o; o += SZ_SD1;
  // reuse of dead regions:
  float* TKR = XQ;    // [B,SN,D1], XQ dead after the 4 compress branches
  float* VZf = XKV;   // [B,SN,D3], XKV dead after the 4 compress branches
  float* MH  = TQZ;   // [B,SN,MLP] (25.2M floats) over dead TQZ..KR span (33.5M)

  auto NT = [&](const float* A, const float* W, float* C, int M, int N, int K) {
    dim3 g((N + 255) / 256, M / 32, Bn);
    gemm_nt_wmma<<<g, 128, 0, stream>>>(A, W, C, M, N, K, (long)M * K, (long)M * N);
  };
  auto NN = [&](const float* Wt, const float* Xb, float* C, int M, int N, int K) {
    dim3 g(N / 64, M / 32, Bn);
    gemm_nn_wmma<<<g, 32, 0, stream>>>(Wt, Xb, C, M, N, K, (long)K * N, (long)M * N);
  };
  auto EW = [&](long n) { return dim3((unsigned)((n + 255) / 256)); };

  // 1) LayerNorms
  ln_kernel<<<dim3(Bn * Ssz), 256, 0, stream>>>(in_q, ln_q_w, XQ, D1n);
  ln_kernel<<<dim3(Bn * Ssz), 256, 0, stream>>>(in_q, ln_kv_w, XKV, D1n);

  // 2) latent mean/var branches: tlin (S->SR) then D1->D2->MVH
  const float* srcs[4]  = {XQ, XQ, XKV, XKV};
  const float* wts[4]   = {Wt_mq, Wt_vq, Wt_mkv, Wt_vkv};
  const float* w1s[4]   = {mq_w1, vq_w1, mkv_w1, vkv_w1};
  const float* w2s[4]   = {mq_w2, vq_w2, mkv_w2, vkv_w2};
  float*       dsts[4]  = {MZQ, VZQ, MZKV, VZKV};
  for (int i = 0; i < 4; ++i) {
    NN(wts[i], srcs[i], T, SRn, D1n, Ssz);
    NT(T, w1s[i], U, SRn, D2n, D1n);
    NT(U, w2s[i], dsts[i], SRn, MVHn, D2n);
  }

  // 3) reparameterize
  long nz = (long)Bn * SRn * MVHn;
  reparam_kernel<<<EW(nz), 256, 0, stream>>>(MZQ, VZQ, eps_zq, dirty_zq, e2de, ZQ, nz, SRn * MVHn);
  reparam_kernel<<<EW(nz), 256, 0, stream>>>(MZKV, VZKV, eps_zkv, dirty_zkv, e2de, ZKV, nz, SRn * MVHn);

  // 4) seq upsample SR->SN (and the S->SN mix for kr)
  NN(Wt_qzu, ZQ,  TQZ, SNn, MVHn, SRn);
  NN(Wt_kzu, ZKV, TKZ, SNn, MVHn, SRn);
  NN(Wt_vzu, ZKV, TVZ, SNn, MVHn, SRn);
  NN(Wt_qr,  ZQ,  TQR, SNn, MVHn, SRn);
  NN(Wt_kr,  in_q, TKR, SNn, D1n, Ssz);

  // 5) feature projections MVH->D2 (and rotary projections)
  NT(TQZ, qz_up, QZ, SNn, D2n, MVHn);
  NT(TKZ, kz_up, KZ, SNn, D2n, MVHn);
  NT(TVZ, vz_up, VZ, SNn, D2n, MVHn);
  NT(TQR, qr_w,  QR, SNn, HHC, MVHn);
  NT(TKR, kr_w,  KR, SNn, HHC, D1n);

  // 6) QKV projections
  NT(QZ, q_w, QC,  SNn, HHC, D2n);
  NT(KZ, k_w, KC,  SNn, HHC, D2n);
  NT(VZ, v_w, VZf, SNn, D3n, D2n);

  // 7) RoPE + head pack -> [B,H,SN,64]
  long nh = (long)Bn * Hn * SNn * HDn;
  rope_pack_kernel<<<EW(nh), 256, 0, stream>>>(QC, QR, rope_qf, Qh);
  rope_pack_kernel<<<EW(nh), 256, 0, stream>>>(KC, KR, rope_kf, Kh);

  // 8) fused attention, O in [B,SN,768]
  attn_kernel<<<dim3(SNn / 16, Hn, Bn), 32, 0, stream>>>(Qh, Kh, VZf, Obuf);

  // 9) output projection + residual
  NT(Obuf, out_w, X, SNn, D3n, D3n);
  long nd = (long)Bn * SNn * D3n;
  addto_kernel<<<EW(nd), 256, 0, stream>>>(X, in_q, nd);

  // 10) LN2 + MLP + final residual
  ln_kernel<<<dim3(Bn * SNn), 256, 0, stream>>>(X, ln2_w, Y, D3n);
  NT(Y, mlp_w1, MH, SNn, MLPn, D3n);
  long nm = (long)Bn * SNn * MLPn;
  gelu_kernel<<<EW(nm), 256, 0, stream>>>(MH, nm);
  NT(MH, mlp_w2, out, SNn, D3n, MLPn);
  addto_kernel<<<EW(nd), 256, 0, stream>>>(out, X, nd);
}
